// LearnableVQ_33414845563603
// MI455X (gfx1250) — matrix operand order repaired and checked by reading, not compiled, not run on hardware
//
#include <hip/hip_runtime.h>
#include <stdint.h>

typedef __attribute__((ext_vector_type(16))) _Float16 v16h;
typedef __attribute__((ext_vector_type(8)))  float    v8f;
typedef __attribute__((ext_vector_type(4)))  float    f32x4;
typedef __attribute__((ext_vector_type(4)))  unsigned int u32x4;
typedef __attribute__((ext_vector_type(8)))  int      i32x8;
typedef __attribute__((ext_vector_type(4)))  int      i32x4;

#define VQ_GAMMA 0.99f
#define VQ_EPS   0.01f

static constexpr int B_ = 8, H_ = 8, L_ = 2048, D_ = 128, S_ = 512;
static constexpr int LTILE = 128;   // L-rows per block
static constexpr int NT    = 16;    // rows per wave

// ---------------------------------------------------------------- init
__global__ void vq_init_kernel(float* accum) {
    if (threadIdx.x < 4) accum[threadIdx.x] = 0.0f;
}

// ---------------------------------------------------------------- prep:
// c = c_sum / max(c_count, EPS); emit c(f32), c(f16), csq; accumulate
// sum(c_count^2) + sum(c_sum^2) into accum[0].
__global__ void __launch_bounds__(128)
vq_prep_kernel(const float* __restrict__ c_sum, const float* __restrict__ c_count,
               float* __restrict__ cb32, _Float16* __restrict__ cb16,
               float* __restrict__ csq, float* __restrict__ accum) {
    int row = blockIdx.x;            // h*S + s
    int d   = threadIdx.x;           // 0..127
    float cntraw = c_count[row];
    float cl     = fmaxf(cntraw, VQ_EPS);
    float cs     = c_sum[(size_t)row * D_ + d];
    float c      = cs / cl;
    cb32[(size_t)row * D_ + d] = c;
    cb16[(size_t)row * D_ + d] = (_Float16)c;

    __shared__ float r1[128];
    __shared__ float r2[128];
    r1[d] = c * c;
    r2[d] = cs * cs;
    __syncthreads();
    for (int s = 64; s > 0; s >>= 1) {
        if (d < s) { r1[d] += r1[d + s]; r2[d] += r2[d + s]; }
        __syncthreads();
    }
    if (d == 0) {
        csq[row] = r1[0];
        atomicAdd(accum + 0, r2[0] + cntraw * cntraw);
    }
}

// ---------------------------------------------------------------- main
__global__ void __launch_bounds__(256, 1)
vq_main_kernel(const float* __restrict__ vecs, const float* __restrict__ c_count,
               const int* __restrict__ loss_mask,
               const float* __restrict__ cb32, const _Float16* __restrict__ cb16_g,
               const float* __restrict__ csq_g,
               float* __restrict__ vecs_hat, float* __restrict__ accum) {
    extern __shared__ char smem[];
    _Float16* lds_cb  = (_Float16*)smem;                     // [S][D] f16 = 128 KB
    float*    lds_csq = (float*)(smem + (size_t)S_ * D_ * 2);// [S]
    float*    lds_vsq = lds_csq + S_;                        // [LTILE]
    float*    lds_bv  = lds_vsq + LTILE;                     // [LTILE]
    int*      lds_bi  = (int*)(lds_bv + LTILE);              // [LTILE]

    const int blk  = blockIdx.x;
    const int tile = blk & 15;
    const int bh   = blk >> 4;
    const int h    = bh & (H_ - 1);
    const int b    = bh >> 3;
    const int l0   = tile * LTILE;
    const int tid  = threadIdx.x;
    const int wave = tid >> 5;
    const int lane = tid & 31;

    // ---- Stage this head's f16 codebook [S=512, D=128] into LDS via TDM.
    if (wave == 0) {
        uint64_t ga = (uint64_t)(uintptr_t)(cb16_g + (size_t)h * S_ * D_);
        uint32_t lds_addr = (uint32_t)(uintptr_t)lds_cb;     // low 32 bits = LDS byte offset
        u32x4 g0;
        g0[0] = 1u;                                          // count=1, user descriptor
        g0[1] = lds_addr;                                    // lds_addr
        g0[2] = (uint32_t)ga;                                // global_addr[31:0]
        g0[3] = (uint32_t)((ga >> 32) & 0x01FFFFFFu) | (2u << 30); // addr[56:32] | type=2
        i32x8 g1;
        g1[0] = (1 << 16);                                   // data_size=2 bytes
        g1[1] = (int)(D_ << 16);                             // tensor_dim0 = 128 (bits 63:48)
        g1[2] = (int)(S_ << 16);                             // tensor_dim1 = 512 (bits 95:80)
        g1[3] = (int)(D_ << 16);                             // tile_dim0 = 128 (bits 127:112)
        g1[4] = S_;                                          // tile_dim1 = 512
        g1[5] = D_;                                          // tensor_dim0_stride = 128
        g1[6] = 0;
        g1[7] = 1;                                           // tensor_dim1_stride = 65536 (hi bits)
        i32x4 g2 = {0, 0, 0, 0};
        i32x4 g3 = {0, 0, 0, 0};
        i32x8 g4 = {0, 0, 0, 0, 0, 0, 0, 0};                 // unused (2D tile)
        __builtin_amdgcn_tensor_load_to_lds(g0, g1, g2, g3, g4, 0);
        __builtin_amdgcn_s_wait_tensorcnt(0);
    }
    // ---- preload csq slice into LDS
    for (int i = tid; i < S_; i += 256) lds_csq[i] = csq_g[(size_t)h * S_ + i];
    __syncthreads();

    // ---- Load A-tile (16 rows x D=128) in WMMA 16-bit A layout, compute vsq.
    const int mrow = wave * NT + (lane & 15);
    const int lrow = l0 + mrow;
    const float* vrow = vecs + (((size_t)(b * H_ + h)) * L_ + lrow) * D_;
    const int boff  = (lane & 16) ? 16 : 0;   // K offset within 32-chunk halves
    v16h a[4];
    float vsq_part = 0.0f;
#pragma unroll
    for (int kc = 0; kc < 4; ++kc) {
        const float* p = vrow + kc * 32 + boff;
        f32x4 f0 = *(const f32x4*)(p + 0);
        f32x4 f1 = *(const f32x4*)(p + 4);
        f32x4 f2 = *(const f32x4*)(p + 8);
        f32x4 f3 = *(const f32x4*)(p + 12);
        v16h av;
        av[0]  = (_Float16)f0.x; av[1]  = (_Float16)f0.y; av[2]  = (_Float16)f0.z; av[3]  = (_Float16)f0.w;
        av[4]  = (_Float16)f1.x; av[5]  = (_Float16)f1.y; av[6]  = (_Float16)f1.z; av[7]  = (_Float16)f1.w;
        av[8]  = (_Float16)f2.x; av[9]  = (_Float16)f2.y; av[10] = (_Float16)f2.z; av[11] = (_Float16)f2.w;
        av[12] = (_Float16)f3.x; av[13] = (_Float16)f3.y; av[14] = (_Float16)f3.z; av[15] = (_Float16)f3.w;
        a[kc] = av;
        vsq_part += f0.x*f0.x + f0.y*f0.y + f0.z*f0.z + f0.w*f0.w;
        vsq_part += f1.x*f1.x + f1.y*f1.y + f1.z*f1.z + f1.w*f1.w;
        vsq_part += f2.x*f2.x + f2.y*f2.y + f2.z*f2.z + f2.w*f2.w;
        vsq_part += f3.x*f3.x + f3.y*f3.y + f3.z*f3.z + f3.w*f3.w;
    }
    float vsq = vsq_part + __shfl_xor(vsq_part, 16, 32);  // lane & lane^16 hold same row
    if (lane < 16) lds_vsq[mrow] = vsq;

    // ---- Scan all S=512 codewords: 32 column tiles x (4 WMMAs over K=128).
    float bestv[8];
    int   bidx[8];
#pragma unroll
    for (int i = 0; i < 8; ++i) { bestv[i] = 3.4e38f; bidx[i] = 0; }
    const int ncol  = lane & 15;
    const int koffB = (lane & 16) ? 16 : 0;

    for (int n0 = 0; n0 < S_; n0 += 16) {
        const int s = n0 + ncol;
        v8f acc = {0.f, 0.f, 0.f, 0.f, 0.f, 0.f, 0.f, 0.f};
#pragma unroll
        for (int kc = 0; kc < 4; ++kc) {
            // B(32x16) f16, contiguous k order as in A: lane holds K = kbase..kbase+15 of column s,
            // contiguous in codeword row-major storage -> one 32B LDS read.
            const v16h bb = *(const v16h*)(lds_cb + (size_t)s * D_ + kc * 32 + koffB);
            acc = __builtin_amdgcn_wmma_f32_16x16x32_f16(
                false, a[kc], false, bb, (short)0, acc, false, false);
        }
        const float cs = lds_csq[s];
#pragma unroll
        for (int i = 0; i < 8; ++i) {
            float cand = fmaf(-2.0f, acc[i], cs);    // csq - 2 v.c  (row-const vsq dropped)
            if (cand < bestv[i]) { bestv[i] = cand; bidx[i] = s; }
        }
    }

    // ---- Reduce min across the 16 lanes of each half (N dimension).
#pragma unroll
    for (int i = 0; i < 8; ++i) {
        float v = bestv[i];
        int   idx = bidx[i];
        for (int off = 8; off >= 1; off >>= 1) {
            float ov  = __shfl_xor(v, off, 16);
            int   oid = __shfl_xor(idx, off, 16);
            if (ov < v || (ov == v && oid < idx)) { v = ov; idx = oid; }
        }
        if ((lane & 15) == 0) {
            int row = wave * NT + i + ((lane & 16) ? 8 : 0);   // C layout M mapping
            lds_bv[row] = v;
            lds_bi[row] = idx;
        }
    }
    __syncthreads();

    // ---- Epilogue: gather f32 codeword, write vecs_hat, accumulate losses.
    float commit = 0.f, cdot = 0.f, sdot = 0.f;
#pragma unroll 4
    for (int r = 0; r < NT; ++r) {
        const int row  = wave * NT + r;
        const int l    = l0 + row;
        const int sidx = lds_bi[row];
        const f32x4* crow = (const f32x4*)(cb32 + ((size_t)h * S_ + sidx) * D_);
        const f32x4* vr   = (const f32x4*)(vecs + (((size_t)(b * H_ + h)) * L_ + l) * D_);
        f32x4* op = (f32x4*)(vecs_hat + (((size_t)(b * H_ + h)) * L_ + l) * D_);
        op[lane] = vr[lane] + crow[lane];
        if (lane == 0) {
            float bval = lds_bv[row];
            float vsqr = lds_vsq[row];
            float lm   = (float)loss_mask[b * L_ + l];
            float err  = fmaxf(vsqr + bval, 0.0f);
            commit += lm * err;
            float cntraw = c_count[(size_t)h * S_ + sidx];
            cdot += lm * cntraw;
            float vdotc = 0.5f * (lds_csq[sidx] - bval);      // v . c
            sdot += lm * fmaxf(cntraw, VQ_EPS) * vdotc;       // v . c_sum
        }
    }
    if (lane == 0) {
        atomicAdd(accum + 1, commit);
        atomicAdd(accum + 2, cdot);
        atomicAdd(accum + 3, sdot);
    }
}

// ---------------------------------------------------------------- finalize
__global__ void vq_final_kernel(const float* __restrict__ accum,
                                const int* __restrict__ nd, const int* __restrict__ nb,
                                float* __restrict__ out_scalars) {
    float scale = (float)(nd[0] * nb[0]);
    out_scalars[0] = accum[1] / (float)(B_ * L_);                               // l_commit
    out_scalars[1] = (1.0f - VQ_GAMMA) * (accum[0] - scale * (accum[2] + accum[3])); // l_codebook
}

// ---------------------------------------------------------------- launch
extern "C" void kernel_launch(void* const* d_in, const int* in_sizes, int n_in,
                              void* d_out, int out_size, void* d_ws, size_t ws_size,
                              hipStream_t stream) {
    const float* vecs      = (const float*)d_in[0];
    const float* c_sum     = (const float*)d_in[1];
    const float* c_count   = (const float*)d_in[2];
    const int*   loss_mask = (const int*)d_in[3];
    const int*   n_device  = (const int*)d_in[4];
    const int*   n_block   = (const int*)d_in[5];
    float* out = (float*)d_out;

    char* ws = (char*)d_ws;
    float*    accum = (float*)ws;                                     // 4 floats
    float*    cb32  = (float*)(ws + 256);                             // H*S*D f32
    _Float16* cb16  = (_Float16*)(ws + 256 + (size_t)H_*S_*D_*4);     // H*S*D f16
    float*    csq   = (float*)(ws + 256 + (size_t)H_*S_*D_*4 + (size_t)H_*S_*D_*2); // H*S f32

    vq_init_kernel<<<1, 32, 0, stream>>>(accum);
    vq_prep_kernel<<<H_ * S_, 128, 0, stream>>>(c_sum, c_count, cb32, cb16, csq, accum);

    size_t shmem = (size_t)S_ * D_ * 2 + (size_t)S_ * 4 + (size_t)LTILE * 12;
    (void)hipFuncSetAttribute((const void*)vq_main_kernel,
                              hipFuncAttributeMaxDynamicSharedMemorySize, (int)shmem);
    vq_main_kernel<<<B_ * H_ * (L_ / LTILE), 256, shmem, stream>>>(
        vecs, c_count, loss_mask, cb32, cb16, csq, out, accum);

    vq_final_kernel<<<1, 1, 0, stream>>>(accum, n_device, n_block,
                                         out + (size_t)B_ * H_ * L_ * D_);
}
